// FasterRCNNModel_75788992905336
// MI455X (gfx1250) — compile-verified, matrix operand order checked by reading, not run on hardware
//
#include <hip/hip_runtime.h>

// ---------------------------------------------------------------------------
// MI455X (gfx1250, wave32) Faster-RCNN forward.
// All conv / FC layers -> bf16 implicit-GEMM on v_wmma_f32_16x16x32_bf16.
// A and B operands are pre-swizzled in global memory into 64x32 WMMA tile
// blocks so GEMM tile fills are contiguous 4KB copies done with gfx1250
// GLOBAL_LOAD_ASYNC_TO_LDS_B128, and every fragment load is two ds_load_b128.
// Feature maps stored HWC so GEMM [M=H*W, N=C] output chains directly.
// ---------------------------------------------------------------------------

typedef __bf16 bf16_t;
typedef bf16_t v16bf __attribute__((ext_vector_type(16)));
typedef float  v8f   __attribute__((ext_vector_type(8)));
typedef int    v4i   __attribute__((ext_vector_type(4)));

#if defined(__HIP_DEVICE_COMPILE__) && __has_builtin(__builtin_amdgcn_global_load_async_to_lds_b128)
#define HAVE_ASYNC_LDS 1
typedef __attribute__((address_space(1))) v4i* gv4i_p;   // global v4i*
typedef __attribute__((address_space(3))) v4i* lv4i_p;   // LDS v4i*
#else
#define HAVE_ASYNC_LDS 0
#endif

// Swizzled operand layout (both A and B):
//   A: [mb = m/64][kb = k/32][mr = m%64][kk = k%32]   (2048 bf16 = 4KB per block)
//   B: [kb = k/32][n (0..Np)][kk = k%32]              (row n holds 32 k-values)
// In both cases the 64x32 tile needed by one workgroup k-step is contiguous.

// ------------------------------ GEMM (WMMA) --------------------------------
__device__ __forceinline__ void store_c(float* C, int m, int n, int M, int N,
                                        int ldc, const float* bias, int relu, float v) {
  if (m < M && n < N) {
    if (bias) v += bias[n];
    if (relu) v = fmaxf(v, 0.0f);
    C[(size_t)m * ldc + n] = v;
  }
}

__global__ __launch_bounds__(128)
void k_gemm(const bf16_t* __restrict__ A, const bf16_t* __restrict__ B,
            float* __restrict__ C, int M, int N, int Np, int Kp,
            const float* __restrict__ bias, int relu, int ldc) {
  // padded rows: 80B stride -> 16 lanes of a b128 frag read hit distinct banks
  __shared__ bf16_t As[64][40];
  __shared__ bf16_t Bs[64][40];

  const int tid  = threadIdx.x;
  const int wave = tid >> 5;
  const int lane = tid & 31;
  const int lm   = lane & 15;
  const int hl   = lane >> 4;          // lane half (wave32)
  const int wm   = (wave >> 1) * 32;   // wave's M sub-offset (0/32)
  const int wn   = (wave & 1) * 32;    // wave's N sub-offset (0/32)
  const int m0   = blockIdx.x * 64;
  const int n0   = blockIdx.y * 64;
  const int nkb  = Kp >> 5;
  const int row  = tid >> 1;           // 0..63 : tile row handled by this thread
  const int colh = (tid & 1) * 16;     // element offset 0/16 (32B half-row)

  const bf16_t* Abase = A + (size_t)blockIdx.x * nkb * 2048;

  v8f acc00 = {}; v8f acc01 = {}; v8f acc10 = {}; v8f acc11 = {};

  for (int kb = 0; kb < nkb; ++kb) {
    const bf16_t* Asrc = Abase + (size_t)kb * 2048 + row * 32 + colh;
    const bf16_t* Bsrc = B + ((size_t)kb * Np + n0) * 32 + row * 32 + colh;
    bf16_t* Adst = &As[row][colh];
    bf16_t* Bdst = &Bs[row][colh];
#if HAVE_ASYNC_LDS
    // gfx1250 async global->LDS (ASYNCcnt), 2x b128 per operand per thread
    __builtin_amdgcn_global_load_async_to_lds_b128((gv4i_p)Asrc,       (lv4i_p)Adst,       0, 0);
    __builtin_amdgcn_global_load_async_to_lds_b128((gv4i_p)(Asrc + 8), (lv4i_p)(Adst + 8), 0, 0);
    __builtin_amdgcn_global_load_async_to_lds_b128((gv4i_p)Bsrc,       (lv4i_p)Bdst,       0, 0);
    __builtin_amdgcn_global_load_async_to_lds_b128((gv4i_p)(Bsrc + 8), (lv4i_p)(Bdst + 8), 0, 0);
#if __has_builtin(__builtin_amdgcn_s_wait_asynccnt)
    __builtin_amdgcn_s_wait_asynccnt(0);
#else
    asm volatile("s_wait_asynccnt 0x0" ::: "memory");
#endif
#else
    *(uint4*)Adst       = *(const uint4*)Asrc;
    *(uint4*)(Adst + 8) = *(const uint4*)(Asrc + 8);
    *(uint4*)Bdst       = *(const uint4*)Bsrc;
    *(uint4*)(Bdst + 8) = *(const uint4*)(Bsrc + 8);
#endif
    if (kb + 1 < nkb) {
      __builtin_prefetch(Asrc + 2048, 0, 0);                    // global_prefetch_b8
      __builtin_prefetch(Bsrc + (size_t)Np * 32, 0, 0);
    }
    __syncthreads();

    // fragments: element e -> k = (e<8) ? hl*8+e : 16+hl*8+(e-8)
    // => exactly two aligned b128 LDS reads per fragment
    union frag { v16bf v; uint4 q[2]; };
    frag a0, a1, b0, b1;
    a0.q[0] = *(const uint4*)&As[wm + lm][hl * 8];
    a0.q[1] = *(const uint4*)&As[wm + lm][16 + hl * 8];
    a1.q[0] = *(const uint4*)&As[wm + 16 + lm][hl * 8];
    a1.q[1] = *(const uint4*)&As[wm + 16 + lm][16 + hl * 8];
    b0.q[0] = *(const uint4*)&Bs[wn + lm][hl * 8];
    b0.q[1] = *(const uint4*)&Bs[wn + lm][16 + hl * 8];
    b1.q[0] = *(const uint4*)&Bs[wn + 16 + lm][hl * 8];
    b1.q[1] = *(const uint4*)&Bs[wn + 16 + lm][16 + hl * 8];

    acc00 = __builtin_amdgcn_wmma_f32_16x16x32_bf16(false, a0.v, false, b0.v, (short)0, acc00, false, false);
    acc01 = __builtin_amdgcn_wmma_f32_16x16x32_bf16(false, a0.v, false, b1.v, (short)0, acc01, false, false);
    acc10 = __builtin_amdgcn_wmma_f32_16x16x32_bf16(false, a1.v, false, b0.v, (short)0, acc10, false, false);
    acc11 = __builtin_amdgcn_wmma_f32_16x16x32_bf16(false, a1.v, false, b1.v, (short)0, acc11, false, false);
    __syncthreads();
  }

  // C layout: VGPR r, lanes 0-15: M=r, lanes 16-31: M=r+8; N=lane%16
  const int nlo = n0 + wn + lm;
  const int nhi = nlo + 16;
#pragma unroll
  for (int r = 0; r < 8; ++r) {
    int mlo = m0 + wm + r + 8 * hl;
    int mhi = mlo + 16;
    store_c(C, mlo, nlo, M, N, ldc, bias, relu, acc00[r]);
    store_c(C, mlo, nhi, M, N, ldc, bias, relu, acc01[r]);
    store_c(C, mhi, nlo, M, N, ldc, bias, relu, acc10[r]);
    store_c(C, mhi, nhi, M, N, ldc, bias, relu, acc11[r]);
  }
}

// ----------------------- data movement / conversion ------------------------
// im2col -> swizzled A blocks. M = Ho*Wo must be a multiple of 64 (true for
// every conv here). Destination-linear indexing => fully coalesced stores.
__global__ void k_im2col(const float* __restrict__ src, bf16_t* __restrict__ dst,
                         int Cin, int H, int W, int KS, int stride, int pad,
                         int Ho, int Wo, int K, int Kp, int chw) {
  int i = blockIdx.x * blockDim.x + threadIdx.x;
  int total = Ho * Wo * Kp;
  if (i >= total) return;
  int nkb = Kp >> 5;
  int kk = i & 31;
  int mr = (i >> 5) & 63;
  int blk = i >> 11;
  int mb = blk / nkb, kb = blk - mb * nkb;
  int m = mb * 64 + mr;
  int k = kb * 32 + kk;
  float v = 0.0f;
  if (k < K) {
    int cin = k / (KS * KS);
    int rem = k - cin * KS * KS;
    int kh = rem / KS, kw = rem - kh * KS;
    int iy = (m / Wo) * stride - pad + kh;
    int ix = (m % Wo) * stride - pad + kw;
    if (iy >= 0 && iy < H && ix >= 0 && ix < W)
      v = chw ? src[(size_t)cin * H * W + iy * W + ix]
              : src[((size_t)iy * W + ix) * Cin + cin];
  }
  dst[i] = (bf16_t)v;
}

// f32 [M][K] row-major -> swizzled A blocks, rows padded to Mp (multiple of 64)
__global__ void k_convert_a(const float* __restrict__ src, bf16_t* __restrict__ dst,
                            int M, int Mp, int K, int Kp) {
  int i = blockIdx.x * blockDim.x + threadIdx.x;
  int total = Mp * Kp;
  if (i >= total) return;
  int nkb = Kp >> 5;
  int kk = i & 31;
  int mr = (i >> 5) & 63;
  int blk = i >> 11;
  int mb = blk / nkb, kb = blk - mb * nkb;
  int m = mb * 64 + mr;
  int k = kb * 32 + kk;
  float v = (m < M && k < K) ? src[(size_t)m * K + k] : 0.0f;
  dst[i] = (bf16_t)v;
}

// weights w[N][K] -> swizzled B: [kb][n][kk]
__global__ void k_convert_w(const float* __restrict__ w, bf16_t* __restrict__ dst,
                            int K, int N, int Kp, int Np) {
  int i = blockIdx.x * blockDim.x + threadIdx.x;
  int total = Kp * Np;
  if (i >= total) return;
  int kk = i & 31;
  int n = (i >> 5) % Np;
  int kb = (i >> 5) / Np;
  int k = kb * 32 + kk;
  float v = (k < K && n < N) ? w[(size_t)n * K + k] : 0.0f;
  dst[i] = (bf16_t)v;
}

// pack cls_w[12,1024] + reg_w[48,1024] -> swizzled B [32 kb][64 n][32 kk]
__global__ void k_pack_clsreg_w(const float* __restrict__ cw, const float* __restrict__ rw,
                                bf16_t* __restrict__ dst) {
  int i = blockIdx.x * blockDim.x + threadIdx.x;
  if (i >= 1024 * 64) return;
  int kk = i & 31;
  int n = (i >> 5) & 63;
  int kb = i >> 11;
  int k = kb * 32 + kk;
  float v = 0.0f;
  if (n < 12) v = cw[(size_t)n * 1024 + k];
  else if (n < 60) v = rw[(size_t)(n - 12) * 1024 + k];
  dst[i] = (bf16_t)v;
}
__global__ void k_pack_clsreg_b(const float* __restrict__ cb, const float* __restrict__ rb,
                                float* __restrict__ dst) {
  int n = threadIdx.x;
  float v = 0.0f;
  if (n < 12) v = cb[n];
  else if (n < 60) v = rb[n - 12];
  dst[n] = v;
}

// --------------------------- elementwise stages ----------------------------
__global__ void k_maxpool(const float* __restrict__ in, float* __restrict__ out,
                          int Hin, int C, int Hout) {
  int i = blockIdx.x * blockDim.x + threadIdx.x;
  int total = Hout * Hout * C;
  if (i >= total) return;
  int c = i % C;
  int x = (i / C) % Hout;
  int y = i / (C * Hout);
  float m = -INFINITY;
  for (int dy = 0; dy < 3; ++dy) {
    int iy = y * 2 + dy;
    if (iy >= Hin) continue;
    for (int dx = 0; dx < 3; ++dx) {
      int ix = x * 2 + dx;
      if (ix >= Hin) continue;
      m = fmaxf(m, in[((size_t)iy * Hin + ix) * C + c]);
    }
  }
  out[i] = m;
}

__global__ void k_add_up2(float* __restrict__ a, const float* __restrict__ b, int W, int C) {
  int i = blockIdx.x * blockDim.x + threadIdx.x;
  int total = W * W * C;
  if (i >= total) return;
  int c = i % C;
  int x = (i / C) % W;
  int y = i / (C * W);
  a[i] += b[(((size_t)(y >> 1)) * (W >> 1) + (x >> 1)) * C + c];
}

// ------------------------ RPN decode / top-k / NMS -------------------------
__global__ void k_decode(const float* __restrict__ logits, const float* __restrict__ deltas,
                         int W, float stride, float size,
                         float* __restrict__ so, float* __restrict__ bo, int n) {
  int i = blockIdx.x * blockDim.x + threadIdx.x;
  if (i >= n) return;
  int m = i / 3, a = i - m * 3;
  int hy = m / W, wx = m - hy * W;
  float ratio = (a == 0) ? 0.5f : ((a == 1) ? 1.0f : 2.0f);
  float hr = sqrtf(ratio);
  float ws = size / hr, hs = size * hr;
  float acx = (wx + 0.5f) * stride, acy = (hy + 0.5f) * stride;
  const float* d = deltas + (size_t)m * 12 + a * 4;
  float dx = d[0], dy = d[1];
  float dw = fminf(fmaxf(d[2], -4.135f), 4.135f);
  float dh = fminf(fmaxf(d[3], -4.135f), 4.135f);
  float cx = dx * ws + acx, cy = dy * hs + acy;
  float w2 = ws * expf(dw) * 0.5f, h2 = hs * expf(dh) * 0.5f;
  so[i] = logits[(size_t)m * 3 + a];
  bo[(size_t)i * 4 + 0] = fminf(fmaxf(cx - w2, 0.0f), 512.0f);
  bo[(size_t)i * 4 + 1] = fminf(fmaxf(cy - h2, 0.0f), 512.0f);
  bo[(size_t)i * 4 + 2] = fminf(fmaxf(cx + w2, 0.0f), 512.0f);
  bo[(size_t)i * 4 + 3] = fminf(fmaxf(cy + h2, 0.0f), 512.0f);
}

__global__ void k_topk(const float* __restrict__ s, const float* __restrict__ b,
                       int n, int k, float* __restrict__ so, float* __restrict__ bo) {
  __shared__ float tile[256];
  int i = blockIdx.x * 256 + threadIdx.x;
  float si = (i < n) ? s[i] : 0.0f;
  int cnt = 0;
  for (int t0 = 0; t0 < n; t0 += 256) {
    int j = t0 + threadIdx.x;
    tile[threadIdx.x] = (j < n) ? s[j] : -INFINITY;
    __syncthreads();
    int lim = min(256, n - t0);
    if (i < n) {
      for (int jj = 0; jj < lim; ++jj) {
        float v = tile[jj];
        cnt += (v > si) || (v == si && (t0 + jj) < i);
      }
    }
    __syncthreads();
  }
  if (i < n && cnt < k) {
    so[cnt] = si;
    bo[(size_t)cnt * 4 + 0] = b[(size_t)i * 4 + 0];
    bo[(size_t)cnt * 4 + 1] = b[(size_t)i * 4 + 1];
    bo[(size_t)cnt * 4 + 2] = b[(size_t)i * 4 + 2];
    bo[(size_t)cnt * 4 + 3] = b[(size_t)i * 4 + 3];
  }
}

__global__ void k_nms(const float* __restrict__ sin, const float* __restrict__ boxes,
                      float* __restrict__ props, int n, int k, float thr) {
  __shared__ float s[4096];
  __shared__ float redv[256];
  __shared__ int   redi[256];
  __shared__ float bx[4];
  int tid = threadIdx.x;
  for (int i = tid; i < n; i += 256) s[i] = sin[i];
  __syncthreads();
  for (int it = 0; it < k; ++it) {
    float best = -INFINITY; int bi = 0;
    for (int i = tid; i < n; i += 256) {
      float v = s[i];
      if (v > best) { best = v; bi = i; }
    }
    redv[tid] = best; redi[tid] = bi;
    __syncthreads();
    for (int off = 128; off > 0; off >>= 1) {
      if (tid < off) {
        if (redv[tid + off] > redv[tid] ||
            (redv[tid + off] == redv[tid] && redi[tid + off] < redi[tid])) {
          redv[tid] = redv[tid + off];
          redi[tid] = redi[tid + off];
        }
      }
      __syncthreads();
    }
    int idx = redi[0];
    if (tid < 4) {
      bx[tid] = boxes[(size_t)idx * 4 + tid];
      props[(size_t)it * 4 + tid] = bx[tid];
    }
    __syncthreads();
    float x1 = bx[0], y1 = bx[1], x2 = bx[2], y2 = bx[3];
    float a = (x2 - x1) * (y2 - y1);
    for (int i = tid; i < n; i += 256) {
      float bx1 = boxes[(size_t)i * 4 + 0], by1 = boxes[(size_t)i * 4 + 1];
      float bx2 = boxes[(size_t)i * 4 + 2], by2 = boxes[(size_t)i * 4 + 3];
      float ix1 = fmaxf(x1, bx1), iy1 = fmaxf(y1, by1);
      float ix2 = fminf(x2, bx2), iy2 = fminf(y2, by2);
      float inter = fmaxf(ix2 - ix1, 0.0f) * fmaxf(iy2 - iy1, 0.0f);
      float bb = (bx2 - bx1) * (by2 - by1);
      float iou = inter / fmaxf(a + bb - inter, 1e-6f);
      if (iou > thr) s[i] = -INFINITY;
    }
    __syncthreads();
  }
}

__global__ void k_roi_levels(const float* __restrict__ props, int* __restrict__ lvls, int n) {
  int r = blockIdx.x * blockDim.x + threadIdx.x;
  if (r >= n) return;
  float w = props[(size_t)r * 4 + 2] - props[(size_t)r * 4 + 0];
  float h = props[(size_t)r * 4 + 3] - props[(size_t)r * 4 + 1];
  float area = fmaxf(w * h, 1e-6f);
  float l = floorf(4.0f + log2f(sqrtf(area) / 224.0f));
  l = fminf(fmaxf(l, 2.0f), 5.0f);
  lvls[r] = (int)l - 2;
}

__global__ __launch_bounds__(256)
void k_roi_align(const float* __restrict__ p2, const float* __restrict__ p3,
                 const float* __restrict__ p4, const float* __restrict__ p5,
                 const float* __restrict__ props, const int* __restrict__ lvls,
                 float* __restrict__ pooled) {
  int r = blockIdx.x;
  int c = threadIdx.x;
  int l = lvls[r];
  const float* f = (l == 0) ? p2 : ((l == 1) ? p3 : ((l == 2) ? p4 : p5));
  int HW = 128 >> l;
  float scale = 0.25f / (float)(1 << l);
  float x1 = props[(size_t)r * 4 + 0] * scale;
  float y1 = props[(size_t)r * 4 + 1] * scale;
  float x2 = props[(size_t)r * 4 + 2] * scale;
  float y2 = props[(size_t)r * 4 + 3] * scale;
  float bw = fmaxf(x2 - x1, 1.0f) * (1.0f / 7.0f);
  float bh = fmaxf(y2 - y1, 1.0f) * (1.0f / 7.0f);
  float acc[49];
#pragma unroll
  for (int j = 0; j < 49; ++j) acc[j] = 0.0f;
  for (int jy = 0; jy < 14; ++jy) {
    float gy = (float)(jy >> 1) + ((float)(jy & 1) + 0.5f) * 0.5f;
    float ys = fminf(fmaxf(y1 + gy * bh, 0.0f), (float)(HW - 1));
    int y0 = (int)floorf(ys);
    int y1i = min(y0 + 1, HW - 1);
    float wy1 = ys - (float)y0, wy0 = 1.0f - wy1;
    for (int jx = 0; jx < 14; ++jx) {
      float gx = (float)(jx >> 1) + ((float)(jx & 1) + 0.5f) * 0.5f;
      float xs = fminf(fmaxf(x1 + gx * bw, 0.0f), (float)(HW - 1));
      int x0 = (int)floorf(xs);
      int x1i = min(x0 + 1, HW - 1);
      float wx1 = xs - (float)x0, wx0 = 1.0f - wx1;
      float v00 = f[((size_t)y0 * HW + x0) * 256 + c];
      float v01 = f[((size_t)y0 * HW + x1i) * 256 + c];
      float v10 = f[((size_t)y1i * HW + x0) * 256 + c];
      float v11 = f[((size_t)y1i * HW + x1i) * 256 + c];
      float v = wy0 * (wx0 * v00 + wx1 * v01) + wy1 * (wx0 * v10 + wx1 * v11);
      acc[(jy >> 1) * 7 + (jx >> 1)] += 0.25f * v;
    }
  }
  float* dst = pooled + ((size_t)r * 256 + c) * 49;
#pragma unroll
  for (int j = 0; j < 49; ++j) dst[j] = acc[j];
}

// -------------------------------- host side --------------------------------
extern "C" void kernel_launch(void* const* d_in, const int* in_sizes, int n_in,
                              void* d_out, int out_size, void* d_ws, size_t ws_size,
                              hipStream_t stream) {
  (void)in_sizes; (void)n_in; (void)out_size; (void)ws_size;
  const float* images = (const float*)d_in[0];
  const float* stem_w = (const float*)d_in[1];  const float* stem_b = (const float*)d_in[2];
  const float* c2w = (const float*)d_in[3];     const float* c2b = (const float*)d_in[4];
  const float* c3w = (const float*)d_in[5];     const float* c3b = (const float*)d_in[6];
  const float* c4w = (const float*)d_in[7];     const float* c4b = (const float*)d_in[8];
  const float* c5w = (const float*)d_in[9];     const float* c5b = (const float*)d_in[10];
  const float* latw[4] = {(const float*)d_in[11], (const float*)d_in[13],
                          (const float*)d_in[15], (const float*)d_in[17]};
  const float* latb[4] = {(const float*)d_in[12], (const float*)d_in[14],
                          (const float*)d_in[16], (const float*)d_in[18]};
  const float* fpnw[4] = {(const float*)d_in[19], (const float*)d_in[21],
                          (const float*)d_in[23], (const float*)d_in[25]};
  const float* fpnb[4] = {(const float*)d_in[20], (const float*)d_in[22],
                          (const float*)d_in[24], (const float*)d_in[26]};
  const float* rpn_w = (const float*)d_in[27];   const float* rpn_b = (const float*)d_in[28];
  const float* rpncls_w = (const float*)d_in[29];const float* rpncls_b = (const float*)d_in[30];
  const float* rpnreg_w = (const float*)d_in[31];const float* rpnreg_b = (const float*)d_in[32];
  const float* fc1_w = (const float*)d_in[33];   const float* fc1_b = (const float*)d_in[34];
  const float* fc2_w = (const float*)d_in[35];   const float* fc2_b = (const float*)d_in[36];
  const float* cls_w = (const float*)d_in[37];   const float* cls_b = (const float*)d_in[38];
  const float* reg_w = (const float*)d_in[39];   const float* reg_b = (const float*)d_in[40];
  float* out = (float*)d_out;

  char* base = (char*)d_ws;
  size_t off = 0;
  auto alloc = [&](size_t bytes) -> void* {
    void* p = base + off;
    off = (off + bytes + 255) & ~(size_t)255;
    return p;
  };

  float* f_stem = (float*)alloc(256ull * 256 * 64 * 4);
  float* f_xp   = (float*)alloc(128ull * 128 * 64 * 4);
  float* f_c2   = (float*)alloc(128ull * 128 * 256 * 4);
  float* f_c3   = (float*)alloc(64ull * 64 * 512 * 4);
  float* f_c4   = (float*)alloc(32ull * 32 * 1024 * 4);
  float* f_c5   = (float*)alloc(16ull * 16 * 2048 * 4);
  float* pm[4], *pf[4];
  int Ws[4] = {128, 64, 32, 16};
  for (int l = 0; l < 4; ++l) {
    pm[l] = (float*)alloc((size_t)Ws[l] * Ws[l] * 256 * 4);
    pf[l] = (float*)alloc((size_t)Ws[l] * Ws[l] * 256 * 4);
  }
  float* tbuf   = (float*)alloc(128ull * 128 * 256 * 4);
  float* logits = (float*)alloc(16384ull * 3 * 4);
  float* deltas = (float*)alloc(16384ull * 12 * 4);
  float* sfull  = (float*)alloc(49152ull * 4);
  float* bfull  = (float*)alloc(49152ull * 4 * 4);
  float* spool  = (float*)alloc(3768ull * 4);
  float* bpool  = (float*)alloc(3768ull * 4 * 4);
  float* props  = (float*)alloc(300ull * 4 * 4);
  int*   lvls   = (int*)alloc(300 * 4);
  float* pooled = (float*)alloc(300ull * 256 * 49 * 4);
  float* h1     = (float*)alloc(300ull * 1024 * 4);
  float* h2     = (float*)alloc(300ull * 1024 * 4);
  float* bias64 = (float*)alloc(64 * 4);
  bf16_t* colbuf = (bf16_t*)alloc(16384ull * 2304 * 2);   // largest swizzled A
  bf16_t* wbuf   = (bf16_t*)alloc(9216ull * 2048 * 2);    // largest swizzled B
  bf16_t* rpn_wb = (bf16_t*)alloc(2304ull * 256 * 2);
  bf16_t* cls_wb = (bf16_t*)alloc(256ull * 64 * 2);
  bf16_t* reg_wb = (bf16_t*)alloc(256ull * 64 * 2);

  auto cdiv = [](int a, int b) { return (a + b - 1) / b; };
  auto convW = [&](const float* w, bf16_t* dst, int K, int N, int Kp, int Np) {
    int tot = Kp * Np;
    k_convert_w<<<cdiv(tot, 256), 256, 0, stream>>>(w, dst, K, N, Kp, Np);
  };
  auto convA = [&](const float* src, int M, int Mp, int K, int Kp) {
    int tot = Mp * Kp;
    k_convert_a<<<cdiv(tot, 256), 256, 0, stream>>>(src, colbuf, M, Mp, K, Kp);
  };
  auto im2col = [&](const float* f, int Cin, int H, int KS, int S, int pad, int Ho,
                    int K, int Kp, int chw) {
    int tot = Ho * Ho * Kp;
    k_im2col<<<cdiv(tot, 256), 256, 0, stream>>>(f, colbuf, Cin, H, H, KS, S, pad,
                                                 Ho, Ho, K, Kp, chw);
  };
  auto gemm = [&](const bf16_t* A, const bf16_t* B, float* C, int M, int N, int Np,
                  int Kp, const float* bias, int relu, int ldc) {
    dim3 g(cdiv(M, 64), Np / 64);
    k_gemm<<<g, 128, 0, stream>>>(A, B, C, M, N, Np, Kp, bias, relu, ldc);
  };

  // ---- backbone ----
  im2col(images, 3, 512, 7, 2, 2, 256, 147, 160, 1);          // stem, CHW input
  convW(stem_w, wbuf, 147, 64, 160, 64);
  gemm(colbuf, wbuf, f_stem, 65536, 64, 64, 160, stem_b, 1, 64);
  k_maxpool<<<cdiv(128 * 128 * 64, 256), 256, 0, stream>>>(f_stem, f_xp, 256, 64, 128);

  im2col(f_xp, 64, 128, 3, 1, 1, 128, 576, 576, 0);
  convW(c2w, wbuf, 576, 256, 576, 256);
  gemm(colbuf, wbuf, f_c2, 16384, 256, 256, 576, c2b, 1, 256);

  im2col(f_c2, 256, 128, 3, 2, 0, 64, 2304, 2304, 0);
  convW(c3w, wbuf, 2304, 512, 2304, 512);
  gemm(colbuf, wbuf, f_c3, 4096, 512, 512, 2304, c3b, 1, 512);

  im2col(f_c3, 512, 64, 3, 2, 0, 32, 4608, 4608, 0);
  convW(c4w, wbuf, 4608, 1024, 4608, 1024);
  gemm(colbuf, wbuf, f_c4, 1024, 1024, 1024, 4608, c4b, 1, 1024);

  im2col(f_c4, 1024, 32, 3, 2, 0, 16, 9216, 9216, 0);
  convW(c5w, wbuf, 9216, 2048, 9216, 2048);
  gemm(colbuf, wbuf, f_c5, 256, 2048, 2048, 9216, c5b, 1, 2048);

  // ---- FPN laterals (1x1) + top-down merge ----
  const float* cfeat[4] = {f_c2, f_c3, f_c4, f_c5};
  int cch[4] = {256, 512, 1024, 2048};
  for (int l = 3; l >= 0; --l) {
    int M = Ws[l] * Ws[l];
    convA(cfeat[l], M, M, cch[l], cch[l]);
    convW(latw[l], wbuf, cch[l], 256, cch[l], 256);
    gemm(colbuf, wbuf, pm[l], M, 256, 256, cch[l], latb[l], 0, 256);
    if (l < 3)
      k_add_up2<<<cdiv(M * 256, 256), 256, 0, stream>>>(pm[l], pm[l + 1], Ws[l], 256);
  }
  // ---- FPN output convs (3x3) ----
  for (int l = 0; l < 4; ++l) {
    int M = Ws[l] * Ws[l];
    im2col(pm[l], 256, Ws[l], 3, 1, 1, Ws[l], 2304, 2304, 0);
    convW(fpnw[l], wbuf, 2304, 256, 2304, 256);
    gemm(colbuf, wbuf, pf[l], M, 256, 256, 2304, fpnb[l], 0, 256);
  }

  // ---- RPN heads + decode + per-level top-k ----
  convW(rpn_w, rpn_wb, 2304, 256, 2304, 256);
  convW(rpncls_w, cls_wb, 256, 3, 256, 64);
  convW(rpnreg_w, reg_wb, 256, 12, 256, 64);
  float strides[4] = {4.f, 8.f, 16.f, 32.f};
  float sizes[4] = {32.f, 64.f, 128.f, 256.f};
  int ks[4] = {1000, 1000, 1000, 768};
  int offs[4] = {0, 1000, 2000, 3000};
  for (int l = 0; l < 4; ++l) {
    int W = Ws[l], M = W * W;
    im2col(pf[l], 256, W, 3, 1, 1, W, 2304, 2304, 0);
    gemm(colbuf, rpn_wb, tbuf, M, 256, 256, 2304, rpn_b, 1, 256);
    convA(tbuf, M, M, 256, 256);
    gemm(colbuf, cls_wb, logits, M, 3, 64, 256, rpncls_b, 0, 3);
    gemm(colbuf, reg_wb, deltas, M, 12, 64, 256, rpnreg_b, 0, 12);
    int Na = M * 3;
    k_decode<<<cdiv(Na, 256), 256, 0, stream>>>(logits, deltas, W, strides[l], sizes[l],
                                                sfull, bfull, Na);
    k_topk<<<cdiv(Na, 256), 256, 0, stream>>>(sfull, bfull, Na, ks[l],
                                              spool + offs[l], bpool + (size_t)offs[l] * 4);
  }

  // ---- NMS + ROIAlign ----
  k_nms<<<1, 256, 0, stream>>>(spool, bpool, props, 3768, 300, 0.7f);
  k_roi_levels<<<2, 256, 0, stream>>>(props, lvls, 300);
  k_roi_align<<<300, 256, 0, stream>>>(pf[0], pf[1], pf[2], pf[3], props, lvls, pooled);

  // ---- box head (FC) ----
  convA(pooled, 300, 320, 12544, 12544);
  convW(fc1_w, wbuf, 12544, 1024, 12544, 1024);
  gemm(colbuf, wbuf, h1, 300, 1024, 1024, 12544, fc1_b, 1, 1024);
  convA(h1, 300, 320, 1024, 1024);
  convW(fc2_w, wbuf, 1024, 1024, 1024, 1024);
  gemm(colbuf, wbuf, h2, 300, 1024, 1024, 1024, fc2_b, 1, 1024);
  convA(h2, 300, 320, 1024, 1024);
  k_pack_clsreg_w<<<cdiv(1024 * 64, 256), 256, 0, stream>>>(cls_w, reg_w, wbuf);
  k_pack_clsreg_b<<<1, 64, 0, stream>>>(cls_b, reg_b, bias64);
  gemm(colbuf, wbuf, out, 300, 60, 64, 1024, bias64, 0, 60);
}
